// GODE_86973087744756
// MI455X (gfx1250) — compile-verified, matrix operand order checked by reading, not run on hardware
//
#include <hip/hip_runtime.h>

// Problem constants (match the reference)
#define Bn    32
#define Nn    2048
#define Dn    128
#define TDn   256
#define HFn   64
#define ROWS  64       // output rows per block
#define KT    32       // K slice per WMMA step
#define DT_STEP 0.01f

typedef __attribute__((ext_vector_type(16))) _Float16 v16h;
typedef __attribute__((ext_vector_type(8)))  _Float16 v8h;
typedef __attribute__((ext_vector_type(8)))  float    v8f;
typedef unsigned int u32x4 __attribute__((ext_vector_type(4)));
typedef int          i32x4 __attribute__((ext_vector_type(4)));
typedef int          i32x8 __attribute__((ext_vector_type(8)));

union V16H { v16h v; v8h h[2]; };

// Load a 16x32 f16 WMMA operand fragment from LDS.
// Works for both A (row-major [m][k]) and B (stored transposed [n][k]):
//   lane 0-15  -> row (lane&15), K = 0..15 ; lane 16-31 -> row (lane&15), K = 16..31
__device__ __forceinline__ v16h ld_frag(const _Float16* p0, int stride, int lane) {
  const _Float16* p = p0 + (lane & 15) * stride + ((lane >> 4) << 4);
  V16H r;
  r.h[0] = *(const v8h*)(p);
  r.h[1] = *(const v8h*)(p + 8);
  return r.v;
}

__device__ __forceinline__ v8f wmma_f16(v16h a, v16h b, v8f c) {
  return __builtin_amdgcn_wmma_f32_16x16x32_f16(false, a, false, b, (short)0, c,
                                                false, false);
}

// Pad control: insert 4 DWORDs (16B) after every 16 DWORDs (64B) -> LDS row
// stride of 40 halves for 32-half tile rows (bank-conflict-avoiding fragments).
#define TDM_PAD64 ((1u << 20) | (3u << 22) | (3u << 25))

// Issue one 2D tensor_load_to_lds: tile_d1 rows of tile_d0 f16 elements,
// global row stride = stride_elems, destination LDS at byte offset lds_addr.
__device__ __forceinline__ void tdm_load_2d(unsigned lds_addr, const void* gptr,
                                            unsigned tensor_d0, unsigned tensor_d1,
                                            unsigned tile_d0,   unsigned tile_d1,
                                            unsigned long long stride_elems,
                                            unsigned padctl) {
  unsigned long long ga = (unsigned long long)(size_t)gptr;   // byte address
  u32x4 g0;
  g0[0] = 1u;                                       // count=1, user descriptor
  g0[1] = lds_addr;                                 // LDS byte address
  g0[2] = (unsigned)ga;                             // global_addr[31:0]
  g0[3] = (unsigned)(ga >> 32) | (2u << 30);        // global_addr[56:32] | type=2
  i32x8 g1;
  g1[0] = (int)((1u << 16) | padctl);               // data_size=1 (2B) + pad ctl
  g1[1] = (int)((tensor_d0 & 0xFFFFu) << 16);       // tensor_dim0[15:0]
  g1[2] = (int)((tensor_d0 >> 16) | ((tensor_d1 & 0xFFFFu) << 16));
  g1[3] = (int)((tensor_d1 >> 16) | (tile_d0 << 16));
  g1[4] = (int)tile_d1;                             // tile_dim1 | tile_dim2=0
  g1[5] = (int)(unsigned)(stride_elems & 0xFFFFFFFFull);  // dim0_stride[31:0]
  g1[6] = (int)(unsigned)(stride_elems >> 32);            // dim0_stride[47:32]
  g1[7] = 0;
  i32x4 z4 = {0, 0, 0, 0};
#if defined(__clang_major__) && (__clang_major__ >= 23)
  i32x8 z8 = {0, 0, 0, 0, 0, 0, 0, 0};
  __builtin_amdgcn_tensor_load_to_lds(g0, g1, z4, z4, z8, 0);
#else
  __builtin_amdgcn_tensor_load_to_lds(g0, g1, z4, z4, 0);
#endif
}

// ---------------------------------------------------------------------------
// Precompute kernels: all WMMA operands staged once as f16.
// ---------------------------------------------------------------------------
__global__ void convert_f16_kernel(const float* __restrict__ in,
                                   _Float16* __restrict__ out, int n) {
  int i = blockIdx.x * 256 + threadIdx.x;
  if (i < n) out[i] = (_Float16)in[i];
}

__global__ void transpose_f16_kernel(const float* __restrict__ in,
                                     _Float16* __restrict__ out, int K, int N) {
  int i = blockIdx.x * 256 + threadIdx.x;
  if (i < K * N) {
    int k = i / N, n = i - k * N;
    out[n * K + k] = (_Float16)in[i];
  }
}

// u16[row][0:128] = x[row], u16[row][128:256] = hz[row]   (row = b*N+n)
__global__ void concat_f16_kernel(const float* __restrict__ x,
                                  const float* __restrict__ hz,
                                  _Float16* __restrict__ u16) {
  int i = blockIdx.x * 256 + threadIdx.x;       // over Bn*Nn*Dn
  int row = i >> 7, col = i & (Dn - 1);
  u16[(size_t)row * TDn + col]      = (_Float16)x[i];
  u16[(size_t)row * TDn + Dn + col] = (_Float16)hz[i];
}

// ---------------------------------------------------------------------------
// Kernel 1: h0 = tanh(relu(u @ Wa + ba) @ Wb + bb); also emits h16^T.
// ---------------------------------------------------------------------------
#define MA_OFF 0                          // u tile   64x40   (2560 halves)
#define MB_OFF 2560                       // W slice  256x40  (10240 halves)
#define MT_OFF 12800                      // relu stage 64x264 (16896 halves)
#define M_TOT  29696

__global__ __launch_bounds__(256, 2)
void mlp_kernel(const _Float16* __restrict__ u16,
                const _Float16* __restrict__ WaT, const float* __restrict__ ba,
                const _Float16* __restrict__ WbT, const float* __restrict__ bb,
                float* __restrict__ hout, _Float16* __restrict__ h16T)
{
  __shared__ __attribute__((aligned(16))) _Float16 smem[M_TOT];

  const int tid  = threadIdx.x;
  const int lane = tid & 31;
  const int wave = tid >> 5;
  const int r    = wave & 3;
  const int cg   = wave >> 2;
  const int row0 = blockIdx.x * ROWS;         // row in [0, B*N)
  const unsigned ldsA_addr = (unsigned)(size_t)&smem[MA_OFF];
  const unsigned ldsB_addr = (unsigned)(size_t)&smem[MB_OFF];

  const v8f vzero = {0.f,0.f,0.f,0.f,0.f,0.f,0.f,0.f};
  v8f acc[8];
  #pragma unroll
  for (int t = 0; t < 8; ++t) acc[t] = vzero;

  // -------- phase 1: T = relu(u @ Wa + ba), N1 = 256, K = 256 --------
  for (int k0 = 0; k0 < TDn; k0 += KT) {
    __syncthreads();
    if (wave == 0) {
      // u tile [64 rows x 32 k], row stride 256
      tdm_load_2d(ldsA_addr, u16 + (size_t)row0 * TDn + k0,
                  TDn, (unsigned)(Bn * Nn), KT, ROWS, TDn, TDM_PAD64);
      // Wa^T slice [256 n][k0..k0+31]
      tdm_load_2d(ldsB_addr, WaT + k0, TDn, TDn, KT, TDn, TDn, TDM_PAD64);
      __builtin_amdgcn_s_wait_tensorcnt(0);
    }
    __syncthreads();

    v16h a = ld_frag(smem + MA_OFF + r * 16 * (KT + 8), KT + 8, lane);
    #pragma unroll
    for (int t = 0; t < 8; ++t) {
      int ct = cg * 8 + t;
      v16h bf = ld_frag(smem + MB_OFF + ct * 16 * (KT + 8), KT + 8, lane);
      acc[t] = wmma_f16(a, bf, acc[t]);
    }
  }
  __syncthreads();

  const int rbase = r * 16 + ((lane >> 4) << 3);
  #pragma unroll
  for (int t = 0; t < 8; ++t) {
    int col = (cg * 8 + t) * 16 + (lane & 15);
    float bav = ba[col];
    #pragma unroll
    for (int j = 0; j < 8; ++j) {
      float v = acc[t][j] + bav;
      smem[MT_OFF + (rbase + j) * (TDn + 8) + col] = (_Float16)fmaxf(v, 0.f);
    }
  }

  // -------- phase 2: h = tanh(T @ Wb + bb), N2 = 128, K = 256 --------
  v8f acc2[4];
  #pragma unroll
  for (int t = 0; t < 4; ++t) acc2[t] = vzero;

  for (int kk = 0; kk < TDn; kk += KT) {
    __syncthreads();
    if (wave == 0) {   // Wb^T slice [128 n][kk..kk+31]
      tdm_load_2d(ldsB_addr, WbT + kk, TDn, Dn, KT, Dn, TDn, TDM_PAD64);
      __builtin_amdgcn_s_wait_tensorcnt(0);
    }
    __syncthreads();

    v16h a2 = ld_frag(smem + MT_OFF + r * 16 * (TDn + 8) + kk, TDn + 8, lane);
    #pragma unroll
    for (int t = 0; t < 4; ++t) {
      int ct = cg * 4 + t;
      v16h b2 = ld_frag(smem + MB_OFF + ct * 16 * (KT + 8), KT + 8, lane);
      acc2[t] = wmma_f16(a2, b2, acc2[t]);
    }
  }

  const int b  = row0 >> 11;            // / Nn
  const int n0 = row0 & (Nn - 1);
  #pragma unroll
  for (int t = 0; t < 4; ++t) {
    int col = (cg * 4 + t) * 16 + (lane & 15);
    float bbv = bb[col];
    v8h pack;
    #pragma unroll
    for (int j = 0; j < 8; ++j) {
      int grow = row0 + rbase + j;
      float h = tanhf(acc2[t][j] + bbv);
      hout[(size_t)grow * Dn + col] = h;
      pack[j] = (_Float16)h;
    }
    *(v8h*)&h16T[((size_t)b * Dn + col) * Nn + n0 + rbase] = pack;
  }
}

// ---------------------------------------------------------------------------
// Kernel 2: one Euler substep.  S = adj@h (K=2048, TDM double-buffered),
// F = tanh(S@W); half-merge epilogue writes f32 h and f16 h^T ping-pong.
// ---------------------------------------------------------------------------
#define GW_OFF 0                          // W^T [n][k]  128x128 (16384 halves)
#define GA_OFF 16384                      // adj tiles x2, 64x40 each (2560)
#define GA_SZ  2560
#define GB_OFF 21504                      // h tiles x2, 128x40 each (5120)
#define GB_SZ  5120
#define GS_OFF 16384                      // S stage 64x136 (8704) aliases tiles
#define G_TOT  31744                      // 63488 bytes

#define NIT (Nn / KT)                     // 64 K-slices

__global__ __launch_bounds__(256, 2)
void gnn_step_kernel(const float* __restrict__ hin, const _Float16* __restrict__ h16in,
                     float* __restrict__ hout, _Float16* __restrict__ h16out,
                     const _Float16* __restrict__ adj16, const _Float16* __restrict__ WT,
                     const float* __restrict__ eps, int which)
{
  __shared__ __attribute__((aligned(16))) _Float16 smem[G_TOT];

  const int tid  = threadIdx.x;
  const int lane = tid & 31;
  const int wave = tid >> 5;
  const int r    = wave & 3;
  const int cg   = wave >> 2;
  const int b    = blockIdx.x >> 5;
  const int row0 = (blockIdx.x & 31) * ROWS;

  const unsigned smem_base = (unsigned)(size_t)&smem[0];
  const _Float16* adjT0 = adj16 + (size_t)row0 * Nn;      // A tile column start
  const _Float16* hT0   = h16in + (size_t)b * Dn * Nn;    // h^T for this batch

  // prologue: W^T (full 128x128, unpadded) + first adj/h tile pair
  if (wave == 0) {
    tdm_load_2d(smem_base + 2u * GW_OFF, WT, Dn, Dn, Dn, Dn, Dn, 0u);
    tdm_load_2d(smem_base + 2u * GA_OFF, adjT0, Nn, Nn, KT, ROWS, Nn, TDM_PAD64);
    tdm_load_2d(smem_base + 2u * GB_OFF, hT0,   Nn, Dn, KT, Dn,   Nn, TDM_PAD64);
  }

  const v8f vzero = {0.f,0.f,0.f,0.f,0.f,0.f,0.f,0.f};
  v8f acc[4];
  #pragma unroll
  for (int t = 0; t < 4; ++t) acc[t] = vzero;

  for (int it = 0; it < NIT; ++it) {
    const int cur = it & 1;
    if (wave == 0) {
      if (it + 1 < NIT) {   // issue next tile pair into the other buffer
        const int nxt = cur ^ 1;
        const int k1  = (it + 1) * KT;
        tdm_load_2d(smem_base + 2u * (GA_OFF + nxt * GA_SZ), adjT0 + k1,
                    Nn, Nn, KT, ROWS, Nn, TDM_PAD64);
        tdm_load_2d(smem_base + 2u * (GB_OFF + nxt * GB_SZ), hT0 + k1,
                    Nn, Dn, KT, Dn, Nn, TDM_PAD64);
        // tensor ops complete in order: <=2 outstanding => current pair landed
        __builtin_amdgcn_s_wait_tensorcnt(2);
      } else {
        __builtin_amdgcn_s_wait_tensorcnt(0);
      }
    }
    __syncthreads();

    v16h a = ld_frag(smem + GA_OFF + cur * GA_SZ + r * 16 * (KT + 8), KT + 8, lane);
    #pragma unroll
    for (int t = 0; t < 4; ++t) {
      int ct = cg * 4 + t;
      v16h bf = ld_frag(smem + GB_OFF + cur * GB_SZ + ct * 16 * (KT + 8), KT + 8, lane);
      acc[t] = wmma_f16(a, bf, acc[t]);
    }
    __syncthreads();   // tile consumed; safe to overwrite on it+1
  }

  // stage S as f16 (aliases the dead tile buffers)
  const int rbase = r * 16 + ((lane >> 4) << 3);
  #pragma unroll
  for (int t = 0; t < 4; ++t) {
    int col = (cg * 4 + t) * 16 + (lane & 15);
    #pragma unroll
    for (int j = 0; j < 8; ++j)
      smem[GS_OFF + (rbase + j) * (Dn + 8) + col] = (_Float16)acc[t][j];
  }
  __syncthreads();

  // -------- phase 2: F = S @ W, K = 128 --------
  v8f acc2[4];
  #pragma unroll
  for (int t = 0; t < 4; ++t) acc2[t] = vzero;
  #pragma unroll
  for (int kk = 0; kk < Dn; kk += KT) {
    v16h a2 = ld_frag(smem + GS_OFF + r * 16 * (Dn + 8) + kk, Dn + 8, lane);
    #pragma unroll
    for (int t = 0; t < 4; ++t) {
      int ct = cg * 4 + t;
      v16h b2 = ld_frag(smem + GW_OFF + ct * 16 * Dn + kk, Dn, lane);
      acc2[t] = wmma_f16(a2, b2, acc2[t]);
    }
  }

  // -------- epilogue: Euler update + half-merge, f32 + f16^T outputs --------
  #pragma unroll
  for (int t = 0; t < 4; ++t) {
    int col = (cg * 4 + t) * 16 + (lane & 15);
    v8h pack;
    #pragma unroll
    for (int j = 0; j < 8; ++j) {
      int grow = row0 + rbase + j;
      size_t g = ((size_t)b * Nn + grow) * Dn + col;
      float hold = hin[g];
      float upd  = hold + DT_STEP * tanhf(acc2[t][j]);
      float o;
      if (which == 0) {
        o = (col < HFn) ? upd : hold;
      } else if (col < HFn) {
        o = hold;
      } else {
        float e = eps[((size_t)b * Nn + grow) * HFn + (col - HFn)];
        e = fminf(fmaxf(e, 0.0f), 0.1f);
        o = e * upd;
      }
      hout[g] = o;
      pack[j] = (_Float16)o;
    }
    *(v8h*)&h16out[((size_t)b * Dn + col) * Nn + row0 + rbase] = pack;
  }
}

// ---------------------------------------------------------------------------
extern "C" void kernel_launch(void* const* d_in, const int* in_sizes, int n_in,
                              void* d_out, int out_size, void* d_ws, size_t ws_size,
                              hipStream_t stream) {
  const float* x   = (const float*)d_in[0];
  const float* hz  = (const float*)d_in[1];
  const float* adj = (const float*)d_in[2];
  const float* W1  = (const float*)d_in[3];
  const float* W2  = (const float*)d_in[4];
  const float* Wa  = (const float*)d_in[5];
  const float* ba  = (const float*)d_in[6];
  const float* Wb  = (const float*)d_in[7];
  const float* bb  = (const float*)d_in[8];
  const float* eps = (const float*)d_in[9];

  float* out = (float*)d_out;                     // [B,N,D] f32

  // workspace carve-up.  u16 (needed only before the first GNN substep)
  // aliases buf32 (first written after the MLP completes) -- same stream,
  // strictly sequential, so no overlap in lifetime.
  char* w = (char*)d_ws;
  float*    buf32 = (float*)w;                                    // 32 MiB
  _Float16* u16   = (_Float16*)w;                                 // 32 MiB alias
  _Float16* adj16 = (_Float16*)(w + (size_t)33554432);            //  8 MiB
  _Float16* h16a  = (_Float16*)(w + (size_t)41943040);            // 16 MiB
  _Float16* h16b  = (_Float16*)(w + (size_t)58720256);            // 16 MiB
  _Float16* W1T   = (_Float16*)(w + (size_t)75497472);
  _Float16* W2T   = (_Float16*)(w + (size_t)75530240);
  _Float16* WaT   = (_Float16*)(w + (size_t)75563008);
  _Float16* WbT   = (_Float16*)(w + (size_t)75694080);

  // one-time f16 operand preparation
  convert_f16_kernel<<<(Nn * Nn) / 256, 256, 0, stream>>>(adj, adj16, Nn * Nn);
  concat_f16_kernel<<<(Bn * Nn * Dn) / 256, 256, 0, stream>>>(x, hz, u16);
  transpose_f16_kernel<<<(Dn * Dn) / 256, 256, 0, stream>>>(W1, W1T, Dn, Dn);
  transpose_f16_kernel<<<(Dn * Dn) / 256, 256, 0, stream>>>(W2, W2T, Dn, Dn);
  transpose_f16_kernel<<<(TDn * TDn) / 256, 256, 0, stream>>>(Wa, WaT, TDn, TDn);
  transpose_f16_kernel<<<(TDn * Dn) / 256, 256, 0, stream>>>(Wb, WbT, TDn, Dn);

  const int mlp_blocks = (Bn * Nn) / ROWS;        // 1024
  const int gnn_blocks = Bn * (Nn / ROWS);        // 1024

  // h0 -> d_out (f32) + h16a (f16^T)
  mlp_kernel<<<mlp_blocks, 256, 0, stream>>>(u16, WaT, ba, WbT, bb, out, h16a);
  // iter 1
  gnn_step_kernel<<<gnn_blocks, 256, 0, stream>>>(out, h16a, buf32, h16b, adj16, W1T, eps, 0);
  gnn_step_kernel<<<gnn_blocks, 256, 0, stream>>>(buf32, h16b, out, h16a, adj16, W2T, eps, 1);
  // iter 2 (final substep writes d_out directly)
  gnn_step_kernel<<<gnn_blocks, 256, 0, stream>>>(out, h16a, buf32, h16b, adj16, W1T, eps, 0);
  gnn_step_kernel<<<gnn_blocks, 256, 0, stream>>>(buf32, h16b, out, h16a, adj16, W2T, eps, 1);
}
